// Refiner_90726889161246
// MI455X (gfx1250) — matrix operand order repaired and checked by reading, not compiled, not run on hardware
//
#include <hip/hip_runtime.h>
#include <hip/hip_bf16.h>
#include <math.h>

typedef __attribute__((ext_vector_type(16))) _Float16 v16h;
typedef __attribute__((ext_vector_type(8)))  _Float16 v8h;
typedef __attribute__((ext_vector_type(4)))  _Float16 v4h;
typedef __attribute__((ext_vector_type(8)))  float    v8f;

#define HID   128
#define INDIM 256

// ---------------- utility kernels ----------------

__global__ void zero_f32(float* __restrict__ p, long n) {
    long i = (long)blockIdx.x * blockDim.x + threadIdx.x;
    if (i < n) p[i] = 0.0f;
}

// D[node] += w[edge], Bdeg[edge] += 1 over all K incidence entries
__global__ void degrees_k(const int* __restrict__ nidx, const int* __restrict__ eidx,
                          const float* __restrict__ w, float* __restrict__ D,
                          float* __restrict__ B, int K) {
    int k = blockIdx.x * blockDim.x + threadIdx.x;
    if (k >= K) return;
    atomicAdd(&D[nidx[k]], w[eidx[k]]);
    atomicAdd(&B[eidx[k]], 1.0f);
}

__global__ void invert_k(float* __restrict__ p, int n) {
    int i = blockIdx.x * blockDim.x + threadIdx.x;
    if (i < n) { float v = p[i]; p[i] = (v > 0.0f) ? (1.0f / v) : 0.0f; }
}

// W [Krows x HID] f32 row-major -> WT [HID x Krows] f16 (i.e. column-major of W)
__global__ void wtrans_f16(const float* __restrict__ W, _Float16* __restrict__ WT, int Krows) {
    int col = blockIdx.x;                         // 0..HID-1
    for (int k = threadIdx.x; k < Krows; k += blockDim.x)
        WT[(long)col * Krows + k] = (_Float16)W[(long)k * HID + col];
}

// plain f32 -> f16 streaming convert (4 elems/thread)
__global__ void cvt_f16(const float* __restrict__ src, _Float16* __restrict__ dst, long n4) {
    long i = (long)blockIdx.x * blockDim.x + threadIdx.x;
    if (i >= n4) return;
    float4 v = ((const float4*)src)[i];
    v4h h;
    h[0] = (_Float16)v.x; h[1] = (_Float16)v.y;
    h[2] = (_Float16)v.z; h[3] = (_Float16)v.w;
    ((v4h*)dst)[i] = h;
}

// fused BatchNorm affine + f16 convert: xh = (f16)(x*scale[f] + shift[f]); HID==128
__global__ void bn_apply_f16(const float* __restrict__ x, const float* __restrict__ scale,
                             const float* __restrict__ shift, _Float16* __restrict__ xh,
                             long n4) {
    long i = (long)blockIdx.x * blockDim.x + threadIdx.x;
    if (i >= n4) return;
    int f = (int)(i & 31) * 4;                    // (i*4) % 128
    float4 v  = ((const float4*)x)[i];
    float4 sc = *(const float4*)(scale + f);
    float4 sh = *(const float4*)(shift + f);
    v4h h;
    h[0] = (_Float16)fmaf(v.x, sc.x, sh.x);
    h[1] = (_Float16)fmaf(v.y, sc.y, sh.y);
    h[2] = (_Float16)fmaf(v.z, sc.z, sh.z);
    h[3] = (_Float16)fmaf(v.w, sc.w, sh.w);
    ((v4h*)xh)[i] = h;
}

// ---------------- WMMA GEMMs (f16 operands, f32 accumulate) ----------------
// One wave = one 16x16 output tile; 8 waves/block cover the 128 output columns.
// A row-major f16 [N x Kdim]; B pre-transposed f16 [HID x Kdim] (column-major of W).
// Inner loop: 4 x global_load_b128 + 1 x v_wmma per K-block of 32.

__device__ __forceinline__ v16h make_a(const _Float16* __restrict__ Arow, int kb, int hi8) {
    v8h alo = *(const v8h*)(Arow + kb + hi8);           // K = kb+hi8   .. +7
    v8h ahi = *(const v8h*)(Arow + kb + 16 + hi8);      // K = kb+16+hi8.. +7
    v16h a;
#pragma unroll
    for (int e = 0; e < 8; ++e) { a[e] = alo[e]; a[e + 8] = ahi[e]; }
    return a;
}

// x = relu(Xh @ Wup + bup); duplicated into x0 (residual identity)
__global__ void linup_wmma(const _Float16* __restrict__ Xh, const _Float16* __restrict__ WT,
                           const float* __restrict__ bias, float* __restrict__ x,
                           float* __restrict__ x0, int Nrows) {
    const int lane  = threadIdx.x & 31;
    const int wave  = threadIdx.x >> 5;
    const int mbase = blockIdx.x * 16;
    const int col   = wave * 16 + (lane & 15);
    const int hi    = (lane & 16);                 // 0 or 16
    const int hi8   = hi >> 1;                     // 0 or 8
    int row = mbase + (lane & 15);
    if (row >= Nrows) row = Nrows - 1;
    const _Float16* Arow = Xh + (long)row * INDIM;
    const _Float16* Bcol = WT + (long)col * INDIM;

    v8f acc = {};
#pragma unroll
    for (int kb = 0; kb < INDIM; kb += 32) {
        v16h a = make_a(Arow, kb, hi8);
        v16h b = make_a(Bcol, kb, hi8);            // same K order as A: 2 x b128
        acc = __builtin_amdgcn_wmma_f32_16x16x32_f16(false, a, false, b,
                                                     (short)0, acc, false, false);
    }
    const float bv = bias[col];
    float* px  = x  + (long)(mbase + hi8) * HID + col;
    float* px0 = x0 + (long)(mbase + hi8) * HID + col;
    if (mbase + 16 <= Nrows) {                     // fast path: whole tile in range
#pragma unroll
        for (int r = 0; r < 8; ++r) {
            float v = fmaxf(acc[r] + bv, 0.0f);
            px[(long)r * HID] = v; px0[(long)r * HID] = v;
        }
    } else {
#pragma unroll
        for (int r = 0; r < 8; ++r) {
            if (mbase + hi8 + r < Nrows) {
                float v = fmaxf(acc[r] + bv, 0.0f);
                px[(long)r * HID] = v; px0[(long)r * HID] = v;
            }
        }
    }
}

// y = xh @ Wc  (BN already folded into xh)
__global__ void conv_wmma(const _Float16* __restrict__ xh, const _Float16* __restrict__ WT,
                          float* __restrict__ y, int Nrows) {
    const int lane  = threadIdx.x & 31;
    const int wave  = threadIdx.x >> 5;
    const int mbase = blockIdx.x * 16;
    const int col   = wave * 16 + (lane & 15);
    const int hi    = (lane & 16);
    const int hi8   = hi >> 1;
    int row = mbase + (lane & 15);
    if (row >= Nrows) row = Nrows - 1;
    const _Float16* Arow = xh + (long)row * HID;
    const _Float16* Bcol = WT + (long)col * HID;

    v8f acc = {};
#pragma unroll
    for (int kb = 0; kb < HID; kb += 32) {
        v16h a = make_a(Arow, kb, hi8);
        v16h b = make_a(Bcol, kb, hi8);
        acc = __builtin_amdgcn_wmma_f32_16x16x32_f16(false, a, false, b,
                                                     (short)0, acc, false, false);
    }
    float* py = y + (long)(mbase + hi8) * HID + col;
    if (mbase + 16 <= Nrows) {
#pragma unroll
        for (int r = 0; r < 8; ++r) py[(long)r * HID] = acc[r];
    } else {
#pragma unroll
        for (int r = 0; r < 8; ++r)
            if (mbase + hi8 + r < Nrows) py[(long)r * HID] = acc[r];
    }
}

// ---------------- BatchNorm stats ----------------

__global__ void bn_zero(float* __restrict__ sums, float* __restrict__ sumsq) {
    int i = threadIdx.x;
    if (i < HID) { sums[i] = 0.0f; sumsq[i] = 0.0f; }
}

__global__ void bn_stats(const float* __restrict__ x, float* __restrict__ sums,
                         float* __restrict__ sumsq, int Nrows) {
    int f = threadIdx.x;                          // 128 threads = 128 features
    float s = 0.0f, s2 = 0.0f;
    for (int r = blockIdx.x; r < Nrows; r += gridDim.x) {
        float v = x[(long)r * HID + f];
        s += v; s2 += v * v;
    }
    atomicAdd(&sums[f], s);
    atomicAdd(&sumsq[f], s2);
}

__global__ void bn_finalize(const float* __restrict__ sums, const float* __restrict__ sumsq,
                            const float* __restrict__ gamma, const float* __restrict__ beta,
                            float* __restrict__ scale, float* __restrict__ shift, int Nrows) {
    int f = threadIdx.x;
    float inv = 1.0f / (float)Nrows;
    float mu  = sums[f] * inv;
    float var = sumsq[f] * inv - mu * mu;
    float rs  = rsqrtf(var + 1e-5f);
    float sc  = rs * gamma[f];
    scale[f] = sc;
    shift[f] = beta[f] - mu * sc;
}

// ---------------- sparse propagation ----------------
// One wave per incidence entry; 32 lanes x float4 cover the 128-float row.

__global__ void scatter_edges(const float* __restrict__ y, const int* __restrict__ nidx,
                              const int* __restrict__ eidx, float* __restrict__ e, int K) {
    int gw   = (int)(((long)blockIdx.x * blockDim.x + threadIdx.x) >> 5);
    int lane = threadIdx.x & 31;
    if (gw >= K) return;
    int n = nidx[gw], ed = eidx[gw];
    const float4 v = ((const float4*)(y + (long)n * HID))[lane];
    float* dst = e + (long)ed * HID + lane * 4;
    atomicAdd(dst + 0, v.x); atomicAdd(dst + 1, v.y);
    atomicAdd(dst + 2, v.z); atomicAdd(dst + 3, v.w);
}

// agg[node] += e[edge] * Binv[edge]  (Binv folded into gather)
__global__ void gather_nodes(const float* __restrict__ e, const int* __restrict__ eidx,
                             const int* __restrict__ nidx, const float* __restrict__ Binv,
                             float* __restrict__ agg, int K) {
    int gw   = (int)(((long)blockIdx.x * blockDim.x + threadIdx.x) >> 5);
    int lane = threadIdx.x & 31;
    if (gw >= K) return;
    int ed = eidx[gw], n = nidx[gw];
    float s = Binv[ed];
    const float4 v = ((const float4*)(e + (long)ed * HID))[lane];
    float* dst = agg + (long)n * HID + lane * 4;
    atomicAdd(dst + 0, v.x * s); atomicAdd(dst + 1, v.y * s);
    atomicAdd(dst + 2, v.z * s); atomicAdd(dst + 3, v.w * s);
}

// x += relu(agg*Dinv + b_conv) * sigmoid(x . w_gate + b_gate)   (wave per node)
__global__ void combine_k(float* __restrict__ x, const float* __restrict__ agg,
                          const float* __restrict__ Dinv, const float* __restrict__ bconv,
                          const float* __restrict__ wgate, const float* __restrict__ bgate,
                          int layer, int Nrows) {
    int gw   = (int)(((long)blockIdx.x * blockDim.x + threadIdx.x) >> 5);
    int lane = threadIdx.x & 31;
    if (gw >= Nrows) return;
    float4 xv = ((const float4*)(x + (long)gw * HID))[lane];
    float4 wv = ((const float4*)wgate)[lane];
    float dot = xv.x * wv.x + xv.y * wv.y + xv.z * wv.z + xv.w * wv.w;
#pragma unroll
    for (int off = 16; off > 0; off >>= 1) dot += __shfl_xor(dot, off, 32);
    float g  = 1.0f / (1.0f + __expf(-(dot + bgate[layer])));
    float di = Dinv[gw];
    float4 av = ((const float4*)(agg + (long)gw * HID))[lane];
    float4 bc = ((const float4*)bconv)[lane];
    xv.x += fmaxf(fmaf(av.x, di, bc.x), 0.0f) * g;
    xv.y += fmaxf(fmaf(av.y, di, bc.y), 0.0f) * g;
    xv.z += fmaxf(fmaf(av.z, di, bc.z), 0.0f) * g;
    xv.w += fmaxf(fmaf(av.w, di, bc.w), 0.0f) * g;
    ((float4*)(x + (long)gw * HID))[lane] = xv;
}

// out = 2*x + identity
__global__ void final_k(const float* __restrict__ x, const float* __restrict__ x0,
                        float* __restrict__ out, long n4) {
    long i = (long)blockIdx.x * blockDim.x + threadIdx.x;
    if (i < n4) {
        float4 a = ((const float4*)x)[i];
        float4 b = ((const float4*)x0)[i];
        float4 o;
        o.x = 2.0f * a.x + b.x; o.y = 2.0f * a.y + b.y;
        o.z = 2.0f * a.z + b.z; o.w = 2.0f * a.w + b.w;
        ((float4*)out)[i] = o;
    }
}

// ---------------- orchestration ----------------

extern "C" void kernel_launch(void* const* d_in, const int* in_sizes, int n_in,
                              void* d_out, int out_size, void* d_ws, size_t ws_size,
                              hipStream_t stream) {
    const float* X     = (const float*)d_in[0];
    const int*   nidx  = (const int*)  d_in[1];
    const int*   eidx  = (const int*)  d_in[2];
    const float* hw    = (const float*)d_in[3];
    const float* Wup   = (const float*)d_in[4];
    const float* bup   = (const float*)d_in[5];
    const float* gamma = (const float*)d_in[6];
    const float* beta  = (const float*)d_in[7];
    const float* Wconv = (const float*)d_in[8];
    const float* bconv = (const float*)d_in[9];
    const float* Wgate = (const float*)d_in[10];
    const float* bgate = (const float*)d_in[11];
    const int N = in_sizes[0] / INDIM;
    const int K = in_sizes[1];
    const int M = in_sizes[3];
    const int L = in_sizes[6] / HID;
    (void)n_in; (void)out_size; (void)ws_size;

    char* wsp = (char*)d_ws;
    auto alloc = [&](size_t bytes) -> char* {
        char* p = wsp; wsp += (bytes + 255) & ~(size_t)255; return p;
    };
    float*    x0     = (float*)alloc((size_t)N * HID * 4);
    float*    x      = (float*)alloc((size_t)N * HID * 4);
    float*    t      = (float*)alloc((size_t)N * HID * 4);   // y, then node aggregate
    float*    e      = (float*)alloc((size_t)M * HID * 4);
    _Float16* Xh     = (_Float16*)alloc((size_t)N * INDIM * 2); // reused as xh [N,HID] f16
    _Float16* WupT   = (_Float16*)alloc((size_t)HID * INDIM * 2);
    _Float16* WconvT = (_Float16*)alloc((size_t)L * HID * HID * 2);
    float*    Dinv   = (float*)alloc((size_t)N * 4);
    float*    Binv   = (float*)alloc((size_t)M * 4);
    float*    sums   = (float*)alloc(HID * 4);
    float*    sumsq  = (float*)alloc(HID * 4);
    float*    scale  = (float*)alloc(HID * 4);
    float*    shift  = (float*)alloc(HID * 4);
    _Float16* xh     = Xh;                                    // alias (Xh dead after lin_up)

    const long NH = (long)N * HID, MH = (long)M * HID;
    const int mtiles = (N + 15) / 16;
    const int entry_blocks = (K + 7) / 8;   // 8 waves/block, 1 entry/wave
    const int node_blocks  = (N + 7) / 8;

    // degrees (layer-invariant)
    zero_f32<<<(N + 255) / 256, 256, 0, stream>>>(Dinv, N);
    zero_f32<<<(M + 255) / 256, 256, 0, stream>>>(Binv, M);
    degrees_k<<<(K + 255) / 256, 256, 0, stream>>>(nidx, eidx, hw, Dinv, Binv, K);
    invert_k<<<(N + 255) / 256, 256, 0, stream>>>(Dinv, N);
    invert_k<<<(M + 255) / 256, 256, 0, stream>>>(Binv, M);

    // weight transposes + X convert (one-time)
    wtrans_f16<<<HID, 256, 0, stream>>>(Wup, WupT, INDIM);
    for (int i = 0; i < L; ++i)
        wtrans_f16<<<HID, 256, 0, stream>>>(Wconv + (size_t)i * HID * HID,
                                            WconvT + (size_t)i * HID * HID, HID);
    long xn4 = (long)N * INDIM / 4;
    cvt_f16<<<(int)((xn4 + 255) / 256), 256, 0, stream>>>(X, Xh, xn4);

    // lin_up + ReLU (WMMA)
    linup_wmma<<<mtiles, 256, 0, stream>>>(Xh, WupT, bup, x, x0, N);

    for (int i = 0; i < L; ++i) {
        bn_zero<<<1, 128, 0, stream>>>(sums, sumsq);
        bn_stats<<<512, 128, 0, stream>>>(x, sums, sumsq, N);
        bn_finalize<<<1, 128, 0, stream>>>(sums, sumsq, gamma + i * HID, beta + i * HID,
                                           scale, shift, N);
        bn_apply_f16<<<(int)((NH / 4 + 255) / 256), 256, 0, stream>>>(x, scale, shift,
                                                                      xh, NH / 4);
        conv_wmma<<<mtiles, 256, 0, stream>>>(xh, WconvT + (size_t)i * HID * HID, t, N);
        zero_f32<<<(int)((MH + 255) / 256), 256, 0, stream>>>(e, MH);
        scatter_edges<<<entry_blocks, 256, 0, stream>>>(t, nidx, eidx, e, K);
        zero_f32<<<(int)((NH + 255) / 256), 256, 0, stream>>>(t, NH);
        gather_nodes<<<entry_blocks, 256, 0, stream>>>(e, eidx, nidx, Binv, t, K);
        combine_k<<<node_blocks, 256, 0, stream>>>(x, t, Dinv, bconv + i * HID,
                                                   Wgate + (size_t)i * HID, bgate, i, N);
    }

    final_k<<<(int)((NH / 4 + 255) / 256), 256, 0, stream>>>(x, x0, (float*)d_out, NH / 4);
}
